// KblamLlamaAttention_27788438405356
// MI455X (gfx1250) — compile-verified, matrix operand order checked by reading, not run on hardware
//
#include <hip/hip_runtime.h>
#include <hip/hip_bf16.h>
#include <math.h>

// ---------------- problem constants ----------------
#define QLEN 2048
#define HEADS 32
#define HDIM 128
#define HID 4096
#define KBLEN 1024
#define TOPK 128
#define NQKV (3 * HID) // 12288
#define SCALE 0.08838834764831845f // 1/sqrt(128)
#define LN_THETA_OVER_HALF 0.14391156509243984f // ln(10000)/64

typedef __attribute__((ext_vector_type(16))) __bf16 v16bf;
typedef __attribute__((ext_vector_type(8)))  __bf16 v8bf;
typedef __attribute__((ext_vector_type(8)))  float  v8f;

// ---------------- async global->LDS (CDNA5), with safe fallback ----------------
#if defined(__has_builtin)
#if __has_builtin(__builtin_amdgcn_global_load_async_to_lds_b128) && \
    __has_builtin(__builtin_amdgcn_s_wait_asynccnt)
#define USE_ASYNC_LDS 1
#endif
#endif

typedef int v4i_vec __attribute__((vector_size(16)));
typedef v4i_vec __attribute__((address_space(1)))* as1_v4i;
typedef v4i_vec __attribute__((address_space(3)))* as3_v4i;

__device__ __forceinline__ void copy16_g2l(const __bf16* __restrict__ g, __bf16* l) {
#ifdef USE_ASYNC_LDS
    __builtin_amdgcn_global_load_async_to_lds_b128((as1_v4i)g, (as3_v4i)l, 0, 0);
#else
    *(uint4*)l = *(const uint4*)g;
#endif
}
// wait until at most N async loads remain outstanding (in-order completion)
template <int N>
__device__ __forceinline__ void wait_async() {
#ifdef USE_ASYNC_LDS
    __builtin_amdgcn_s_wait_asynccnt(N);
#endif
}

// ---------------- small helper kernels ----------------

__global__ void k_colsum(const float* __restrict__ hs, float* __restrict__ out) {
    int d = blockIdx.x * blockDim.x + threadIdx.x;
    if (d >= HID) return;
    float acc = 0.f;
    for (int q = 0; q < QLEN; ++q) acc += hs[(size_t)q * HID + d];
    out[d] = acc;
}

__global__ void k_rowdot(const float* __restrict__ M, const float* __restrict__ vec,
                         float* __restrict__ out, int K) {
    __shared__ float red[256];
    int row = blockIdx.x;
    float p = 0.f;
    for (int i = threadIdx.x; i < K; i += 256)
        p += M[(size_t)row * K + i] * vec[i];
    red[threadIdx.x] = p;
    __syncthreads();
    for (int s = 128; s > 0; s >>= 1) {
        if (threadIdx.x < s) red[threadIdx.x] += red[threadIdx.x + s];
        __syncthreads();
    }
    if (threadIdx.x == 0) out[row] = red[0];
}

__global__ void k_topk(const float* __restrict__ scores, int* __restrict__ idx) {
    __shared__ float sv[KBLEN];
    __shared__ float rv[KBLEN];
    __shared__ int   ri[KBLEN];
    int t = threadIdx.x;
    sv[t] = scores[t];
    __syncthreads();
    for (int k = 0; k < TOPK; ++k) {
        rv[t] = sv[t]; ri[t] = t;
        __syncthreads();
        for (int s = KBLEN / 2; s > 0; s >>= 1) {
            if (t < s) {
                float a = rv[t], b = rv[t + s];
                if (b > a || (b == a && ri[t + s] < ri[t])) { rv[t] = b; ri[t] = ri[t + s]; }
            }
            __syncthreads();
        }
        if (t == 0) { idx[k] = ri[0]; sv[ri[0]] = -1e30f; }
        __syncthreads();
    }
}

__global__ void k_gather(const float* __restrict__ kb_keys, const float* __restrict__ kb_vals,
                         const int* __restrict__ idx,
                         __bf16* __restrict__ ksel, __bf16* __restrict__ vsel) {
    long n = (long)HEADS * TOPK * HDIM;
    for (long e = blockIdx.x * (long)blockDim.x + threadIdx.x; e < n;
         e += (long)gridDim.x * blockDim.x) {
        int d = (int)(e % HDIM);
        int t = (int)((e / HDIM) % TOPK);
        int h = (int)(e / ((long)HDIM * TOPK));
        size_t src = (size_t)idx[t] * HID + h * HDIM + d;
        ksel[e] = (__bf16)kb_keys[src];
        vsel[e] = (__bf16)kb_vals[src];
    }
}

__global__ void k_cvt(const float* __restrict__ s, __bf16* __restrict__ d, long n) {
    for (long i = blockIdx.x * (long)blockDim.x + threadIdx.x; i < n;
         i += (long)gridDim.x * blockDim.x)
        d[i] = (__bf16)s[i];
}

// ---------------- WMMA GEMM (double-buffered async LDS staging) ----------------
// C[M][N] = sum_k A[M][k] * B[N][k]; block tile 128x256, 8 waves, 64x64/wave.
__global__ __launch_bounds__(256) void k_gemm_bf16(
    const __bf16* __restrict__ A, const __bf16* __restrict__ B,
    float* __restrict__ C, int M, int N, int K) {
    __shared__ __attribute__((aligned(16))) __bf16 As[2][128 * 40];
    __shared__ __attribute__((aligned(16))) __bf16 Bs[2][256 * 40];

    const int tid  = threadIdx.x;
    const int lane = tid & 31;
    const int wv   = tid >> 5;
    const int wm   = wv >> 2;
    const int wn   = wv & 3;
    const int l15  = lane & 15;
    const int koff = (lane >> 4) * 8;
    const int aRow0 = blockIdx.y * 128;
    const int bRow0 = blockIdx.x * 256;

    // per-wave: 2 A + 4 B async b128 loads per K-step
    auto issue = [&](int buf, int kb) {
        #pragma unroll
        for (int r = 0; r < 2; ++r) {
            int i   = tid * 2 + r;
            int row = i >> 2, c8 = i & 3;
            copy16_g2l(&A[(size_t)(aRow0 + row) * K + kb + c8 * 8],
                       &As[buf][row * 40 + c8 * 8]);
        }
        #pragma unroll
        for (int r = 0; r < 4; ++r) {
            int i   = tid * 4 + r;
            int row = i >> 2, c8 = i & 3;
            copy16_g2l(&B[(size_t)(bRow0 + row) * K + kb + c8 * 8],
                       &Bs[buf][row * 40 + c8 * 8]);
        }
        if (kb + 32 < K) {
            __builtin_prefetch(&A[(size_t)(aRow0 + (tid >> 1)) * K + kb + 32], 0, 1);
            __builtin_prefetch(&B[(size_t)(bRow0 + tid) * K + kb + 32], 0, 1);
        }
    };

    v8f acc[4][4];
    #pragma unroll
    for (int i = 0; i < 4; ++i)
        #pragma unroll
        for (int j = 0; j < 4; ++j)
            #pragma unroll
            for (int e = 0; e < 8; ++e) acc[i][j][e] = 0.f;

    issue(0, 0);
    const int nk = K >> 5;
    for (int t = 0; t < nk; ++t) {
        const int cur = t & 1;
        if (t + 1 < nk) {
            issue(cur ^ 1, (t + 1) << 5); // prefetch next slab into other buffer
            wait_async<6>();              // drain current tile only (in-order)
        } else {
            wait_async<0>();
        }
        __syncthreads();

        v16bf afrag[4], bfrag[4];
        #pragma unroll
        for (int i = 0; i < 4; ++i) {
            int r = wm * 64 + i * 16 + l15;
            ((v8bf*)&afrag[i])[0] = *(const v8bf*)&As[cur][r * 40 + koff];
            ((v8bf*)&afrag[i])[1] = *(const v8bf*)&As[cur][r * 40 + 16 + koff];
        }
        #pragma unroll
        for (int j = 0; j < 4; ++j) {
            int r = wn * 64 + j * 16 + l15;
            ((v8bf*)&bfrag[j])[0] = *(const v8bf*)&Bs[cur][r * 40 + koff];
            ((v8bf*)&bfrag[j])[1] = *(const v8bf*)&Bs[cur][r * 40 + 16 + koff];
        }
        #pragma unroll
        for (int i = 0; i < 4; ++i)
            #pragma unroll
            for (int j = 0; j < 4; ++j)
                acc[i][j] = __builtin_amdgcn_wmma_f32_16x16x32_bf16(
                    false, afrag[i], false, bfrag[j], (short)0, acc[i][j], false, false);
        __syncthreads();
    }

    #pragma unroll
    for (int i = 0; i < 4; ++i) {
        int rowBase = aRow0 + wm * 64 + i * 16 + (lane >> 4) * 8;
        #pragma unroll
        for (int j = 0; j < 4; ++j) {
            int col = bRow0 + wn * 64 + j * 16 + l15;
            #pragma unroll
            for (int e = 0; e < 8; ++e)
                C[(size_t)(rowBase + e) * N + col] = acc[i][j][e];
        }
    }
}

// ---------------- RoPE + head split ----------------
__global__ void k_rope_split(const float* __restrict__ Cqkv,
                             __bf16* __restrict__ qh, __bf16* __restrict__ kh,
                             __bf16* __restrict__ vh) {
    long n = (long)QLEN * HID;
    for (long e = blockIdx.x * (long)blockDim.x + threadIdx.x; e < n;
         e += (long)gridDim.x * blockDim.x) {
        int c   = (int)(e % HID);
        int pos = (int)(e / HID);
        int h = c / HDIM, d = c % HDIM;
        size_t base = (size_t)pos * NQKV;
        float qv = Cqkv[base + c];
        float kv = Cqkv[base + HID + c];
        float vv = Cqkv[base + 2 * HID + c];
        float invf = __expf(-(float)(d & 63) * LN_THETA_OVER_HALF);
        float ang = (float)pos * invf;
        float cs = __cosf(ang), sn = __sinf(ang);
        int hc = h * HDIM;
        float qro = (d < 64) ? -Cqkv[base + hc + d + 64] : Cqkv[base + hc + d - 64];
        float kro = (d < 64) ? -Cqkv[base + HID + hc + d + 64] : Cqkv[base + HID + hc + d - 64];
        size_t dst = ((size_t)h * QLEN + pos) * HDIM + d;
        qh[dst] = (__bf16)(qv * cs + qro * sn);
        kh[dst] = (__bf16)(kv * cs + kro * sn);
        vh[dst] = (__bf16)vv;
    }
}

// ---------------- WMMA flash attention (K double-buffered async) ----------------
// grid (QLEN/64, HEADS), 4 waves; wave w owns q rows [qBase+16w, qBase+16w+16).
// key tiles of 64: 2 KB tiles (score_shift, unmasked) then causal tiles.
__global__ __launch_bounds__(128) void k_flash(
    const __bf16* __restrict__ qh, const __bf16* __restrict__ kh,
    const __bf16* __restrict__ vh, const __bf16* __restrict__ ksel,
    const __bf16* __restrict__ vsel, const float* __restrict__ score_shift,
    float* __restrict__ O) {
    __shared__ __attribute__((aligned(16))) __bf16 Kt[2][64 * 136]; // [key][dim]
    __shared__ __attribute__((aligned(16))) __bf16 Vt[128 * 72];    // [dim][key]
    __shared__ __attribute__((aligned(16))) __bf16 Ps[4 * 16 * 72]; // per-wave P

    const int h = blockIdx.y;
    const int qtile = blockIdx.x;
    const int qBase = qtile * 64;
    const int tid = threadIdx.x;
    const int lane = tid & 31, wid = tid >> 5;
    const int l15 = lane & 15, half = lane >> 4, koff = half * 8;
    const int rowBase = qBase + wid * 16;

    const __bf16* qhead = qh + (size_t)h * QLEN * HDIM;
    const __bf16* khead = kh + (size_t)h * QLEN * HDIM;
    const __bf16* vhead = vh + (size_t)h * QLEN * HDIM;
    const __bf16* kkb = ksel + (size_t)h * TOPK * HDIM;
    const __bf16* vkb = vsel + (size_t)h * TOPK * HDIM;
    const float shift = score_shift[h];

    // Q fragments (A layout): lane l15 = row rowBase+l15, 4 K-chunks of 32 dims
    v16bf qfrag[4];
    {
        const __bf16* qrow = qhead + (size_t)(rowBase + l15) * HDIM;
        #pragma unroll
        for (int c = 0; c < 4; ++c) {
            ((v8bf*)&qfrag[c])[0] = *(const v8bf*)(qrow + c * 32 + koff);
            ((v8bf*)&qfrag[c])[1] = *(const v8bf*)(qrow + c * 32 + 16 + koff);
        }
    }

    v8f oacc[8];
    #pragma unroll
    for (int n = 0; n < 8; ++n)
        #pragma unroll
        for (int e = 0; e < 8; ++e) oacc[n][e] = 0.f;
    float mrow[8], lrow[8];
    #pragma unroll
    for (int e = 0; e < 8; ++e) { mrow[e] = -1e30f; lrow[e] = 0.f; }

    // 8 async b128 loads per thread per key tile
    auto issueK = [&](int buf, int t) {
        const bool isKB = (t < 2);
        const int keyBase = isKB ? t * 64 : (t - 2) * 64;
        const __bf16* ksrc = (isKB ? kkb : khead) + (size_t)keyBase * HDIM;
        #pragma unroll
        for (int r = 0; r < 8; ++r) {
            int i = tid + 128 * r;          // 0..1023
            int key = i >> 4, c8 = i & 15;
            copy16_g2l(&ksrc[(size_t)key * HDIM + c8 * 8], &Kt[buf][key * 136 + c8 * 8]);
        }
    };

    const int nTiles = 2 + qtile + 1;
    issueK(0, 0);
    for (int t = 0; t < nTiles; ++t) {
        const int cur = t & 1;
        const bool isKB = (t < 2);
        const int keyBase = isKB ? t * 64 : (t - 2) * 64;
        const __bf16* vsrc = (isKB ? vkb : vhead) + (size_t)keyBase * HDIM;

        if (t + 1 < nTiles) issueK(cur ^ 1, t + 1); // next K tile in flight

        // stage V transposed (sync; overlaps with in-flight K asyncs)
        #pragma unroll
        for (int r = 0; r < 8; ++r) {
            int i = tid + 128 * r;
            int key = i >> 4, dg = i & 15;
            v8bf vv = *(const v8bf*)(vsrc + (size_t)key * HDIM + dg * 8);
            #pragma unroll
            for (int e = 0; e < 8; ++e) Vt[(dg * 8 + e) * 72 + key] = vv[e];
        }
        if (t + 1 < nTiles) wait_async<8>(); else wait_async<0>();
        __syncthreads();

        if (isKB || keyBase <= rowBase + 15) {
            // S = Q K^T : 4 n-tiles of 16 keys, K-loop 4x32 dims
            v8f sAcc[4];
            #pragma unroll
            for (int j = 0; j < 4; ++j)
                #pragma unroll
                for (int e = 0; e < 8; ++e) sAcc[j][e] = 0.f;
            #pragma unroll
            for (int c = 0; c < 4; ++c) {
                #pragma unroll
                for (int j = 0; j < 4; ++j) {
                    v16bf bfrag;
                    const __bf16* kr = &Kt[cur][(j * 16 + l15) * 136 + c * 32];
                    ((v8bf*)&bfrag)[0] = *(const v8bf*)(kr + koff);
                    ((v8bf*)&bfrag)[1] = *(const v8bf*)(kr + 16 + koff);
                    sAcc[j] = __builtin_amdgcn_wmma_f32_16x16x32_bf16(
                        false, qfrag[c], false, bfrag, (short)0, sAcc[j], false, false);
                }
            }
            // scale + shift/causal mask
            #pragma unroll
            for (int j = 0; j < 4; ++j) {
                int key = keyBase + j * 16 + l15;
                #pragma unroll
                for (int e = 0; e < 8; ++e) {
                    float s = sAcc[j][e] * SCALE + (isKB ? shift : 0.f);
                    if (!isKB && key > rowBase + e + half * 8) s = -1e30f;
                    sAcc[j][e] = s;
                }
            }
            // online softmax (row = e + 8*half; its 16 cols live in one lane-half)
            float corr[8];
            #pragma unroll
            for (int e = 0; e < 8; ++e) {
                float v = fmaxf(fmaxf(sAcc[0][e], sAcc[1][e]), fmaxf(sAcc[2][e], sAcc[3][e]));
                #pragma unroll
                for (int x = 8; x > 0; x >>= 1) v = fmaxf(v, __shfl_xor(v, x));
                float nm = fmaxf(mrow[e], v);
                corr[e] = __expf(mrow[e] - nm);
                mrow[e] = nm;
            }
            #pragma unroll
            for (int e = 0; e < 8; ++e) {
                float su = 0.f;
                #pragma unroll
                for (int j = 0; j < 4; ++j) {
                    float p = __expf(sAcc[j][e] - mrow[e]);
                    sAcc[j][e] = p;
                    su += p;
                }
                #pragma unroll
                for (int x = 8; x > 0; x >>= 1) su += __shfl_xor(su, x);
                lrow[e] = lrow[e] * corr[e] + su;
            }
            #pragma unroll
            for (int n = 0; n < 8; ++n)
                #pragma unroll
                for (int e = 0; e < 8; ++e) oacc[n][e] *= corr[e];
            // P (C layout) -> LDS -> A layout
            __bf16* pw = &Ps[wid * 16 * 72];
            #pragma unroll
            for (int j = 0; j < 4; ++j) {
                int col = j * 16 + l15;
                #pragma unroll
                for (int e = 0; e < 8; ++e)
                    pw[(e + half * 8) * 72 + col] = (__bf16)sAcc[j][e];
            }
            // O += P V : K-dim 64 keys (2 chunks), 8 n-tiles of 16 dims
            #pragma unroll
            for (int kc = 0; kc < 2; ++kc) {
                v16bf afrag;
                const __bf16* pr = pw + (size_t)l15 * 72 + kc * 32;
                ((v8bf*)&afrag)[0] = *(const v8bf*)(pr + koff);
                ((v8bf*)&afrag)[1] = *(const v8bf*)(pr + 16 + koff);
                #pragma unroll
                for (int n = 0; n < 8; ++n) {
                    v16bf bfrag;
                    const __bf16* vr = &Vt[(n * 16 + l15) * 72 + kc * 32];
                    ((v8bf*)&bfrag)[0] = *(const v8bf*)(vr + koff);
                    ((v8bf*)&bfrag)[1] = *(const v8bf*)(vr + 16 + koff);
                    oacc[n] = __builtin_amdgcn_wmma_f32_16x16x32_bf16(
                        false, afrag, false, bfrag, (short)0, oacc[n], false, false);
                }
            }
        }
        __syncthreads();
    }

    #pragma unroll
    for (int n = 0; n < 8; ++n) {
        #pragma unroll
        for (int e = 0; e < 8; ++e) {
            int q = rowBase + e + half * 8;
            O[(size_t)q * HID + h * HDIM + n * 16 + l15] = oacc[n][e] / lrow[e];
        }
    }
}

// ---------------- host orchestration ----------------
extern "C" void kernel_launch(void* const* d_in, const int* in_sizes, int n_in,
                              void* d_out, int out_size, void* d_ws, size_t ws_size,
                              hipStream_t stream) {
    const float* hs       = (const float*)d_in[0];
    const float* kb_keys  = (const float*)d_in[3];
    const float* kb_vals  = (const float*)d_in[4];
    const float* Wq       = (const float*)d_in[5];
    const float* Wq2      = (const float*)d_in[6];
    const float* Wk       = (const float*)d_in[7];
    const float* Wv       = (const float*)d_in[8];
    const float* Wo       = (const float*)d_in[9];
    const float* shift    = (const float*)d_in[10];
    float* out            = (float*)d_out;

    char* base = (char*)d_ws;
    size_t off = 0;
    auto take = [&](size_t bytes) -> void* {
        off = (off + 255) & ~(size_t)255;
        void* p = base + off;
        off += bytes;
        return p;
    };
    const size_t W_ELEMS = (size_t)HID * HID;
    __bf16* hs_bf   = (__bf16*)take((size_t)QLEN * HID * 2);
    __bf16* Wqkv_bf = (__bf16*)take((size_t)NQKV * HID * 2);
    __bf16* Wo_bf   = (__bf16*)take(W_ELEMS * 2);
    float*  Cqkv    = (float*) take((size_t)QLEN * NQKV * 4);
    __bf16* qh      = (__bf16*)take((size_t)QLEN * HID * 2);
    __bf16* kh      = (__bf16*)take((size_t)QLEN * HID * 2);
    __bf16* vh      = (__bf16*)take((size_t)QLEN * HID * 2);
    __bf16* ksel    = (__bf16*)take((size_t)HEADS * TOPK * HDIM * 2);
    __bf16* vsel    = (__bf16*)take((size_t)HEADS * TOPK * HDIM * 2);
    float*  Obuf    = (float*) take((size_t)QLEN * HID * 4);
    __bf16* O_bf    = (__bf16*)take((size_t)QLEN * HID * 2);
    float*  hs_sum  = (float*) take(HID * 4);
    float*  q2sum   = (float*) take(HID * 4);
    float*  scores  = (float*) take(KBLEN * 4);
    int*    idx     = (int*)   take(TOPK * 4);

    // KB scoring shortcut: scores = ((sum_q hs) @ Wq2^T) . kb_keys
    k_colsum<<<HID / 256, 256, 0, stream>>>(hs, hs_sum);
    k_rowdot<<<HID, 256, 0, stream>>>(Wq2, hs_sum, q2sum, HID);
    k_rowdot<<<KBLEN, 256, 0, stream>>>(kb_keys, q2sum, scores, HID);
    k_topk<<<1, KBLEN, 0, stream>>>(scores, idx);
    k_gather<<<512, 256, 0, stream>>>(kb_keys, kb_vals, idx, ksel, vsel);

    // bf16 conversions; pack Wq|Wk|Wv
    long nact = (long)QLEN * HID;
    k_cvt<<<4096, 256, 0, stream>>>(hs, hs_bf, nact);
    k_cvt<<<4096, 256, 0, stream>>>(Wq, Wqkv_bf, (long)W_ELEMS);
    k_cvt<<<4096, 256, 0, stream>>>(Wk, Wqkv_bf + W_ELEMS, (long)W_ELEMS);
    k_cvt<<<4096, 256, 0, stream>>>(Wv, Wqkv_bf + 2 * W_ELEMS, (long)W_ELEMS);
    k_cvt<<<4096, 256, 0, stream>>>(Wo, Wo_bf, (long)W_ELEMS);

    // fused qkv GEMM
    k_gemm_bf16<<<dim3(NQKV / 256, QLEN / 128), 256, 0, stream>>>(
        hs_bf, Wqkv_bf, Cqkv, QLEN, NQKV, HID);

    // RoPE + head split
    k_rope_split<<<4096, 256, 0, stream>>>(Cqkv, qh, kh, vh);

    // WMMA flash attention
    k_flash<<<dim3(QLEN / 64, HEADS), 128, 0, stream>>>(qh, kh, vh, ksel, vsel, shift, Obuf);

    // output projection
    k_cvt<<<4096, 256, 0, stream>>>(Obuf, O_bf, nact);
    k_gemm_bf16<<<dim3(HID / 256, QLEN / 128), 256, 0, stream>>>(
        O_bf, Wo_bf, out, QLEN, HID, HID);
}
